// ScoreNet_57269093925345
// MI455X (gfx1250) — matrix operand order repaired and checked by reading, not compiled, not run on hardware
//
#include <hip/hip_runtime.h>
#include <hip/hip_bf16.h>
#include <math.h>

#define N_NODES 40962
#define N_EDGES 245772
// MUL_IN=8, MUL_OUT=16, H=64, WNUM=512

typedef __attribute__((ext_vector_type(16))) __bf16 v16bf;
typedef __attribute__((ext_vector_type(8)))  float  v8f;
typedef __attribute__((ext_vector_type(2)))  float  v2f;

// fast silu/sigmoid: v_exp_f32 + v_rcp_f32 (no IEEE div sequences)
__device__ __forceinline__ float silu_f(float x) {
  return x * __builtin_amdgcn_rcpf(1.0f + __expf(-x));
}
__device__ __forceinline__ float sigm_f(float x) {
  return __builtin_amdgcn_rcpf(1.0f + __expf(-x));
}

// ---------------------------------------------------------------------------
// Pre-pack W2 (64x512 f32, row-major) into bf16 in the WMMA B-operand layout.
// B for v_wmma_f32_16x16x32_bf16 is 32x16 (KxN) per tile:
//   lanes 0-15 hold K=0..7,16..23 (element s: K=s for s<8, K=8+s for s>=8), N=lane%16
//   lanes 16-31 hold K=8..15,24..31 likewise.
// Tile linear id = t*2 + kk  (t = column tile 0..31, kk = K-slab 0..1).
// Scale folded in: (1/sqrt(H)) * (1/sqrt(2*MUL_IN)) = (1/8)*(1/4) = 1/32.
// ---------------------------------------------------------------------------
__global__ void prepack_w2(const float* __restrict__ W2, __bf16* __restrict__ Wp) {
  int idx = blockIdx.x * blockDim.x + threadIdx.x;   // 0 .. 64*512-1
  if (idx >= 64 * 512) return;
  int k   = idx >> 9;          // hidden row 0..63
  int col = idx & 511;         // output col 0..511
  int kk  = k >> 5;            // K slab
  int kl  = k & 31;
  int grp = (kl >> 3) & 1;
  int kl2 = (kl & 7) | ((kl >> 4) << 3);
  int v   = kl2 >> 1, j = kl2 & 1;
  int t   = col >> 4, n = col & 15;
  int lane = grp * 16 + n;
  int out  = ((t * 2 + kk) * 32 + lane) * 16 + (v * 2 + j);
  Wp[out] = (__bf16)(W2[idx] * (1.0f / 32.0f));
}

// ---------------------------------------------------------------------------
// Edge kernel: one wave32 per 16-edge tile.
// ---------------------------------------------------------------------------
struct EdgeSmem {
  float len[16];
  float dir[16][3];
  float xs[16][8];
  float dot[16][8];
  float xv[16][8][3];
  int   dst[16];
};

__launch_bounds__(256)
__global__ void edge_kernel(const float* __restrict__ x,
                            const float* __restrict__ pos,
                            const int*   __restrict__ srcI,
                            const int*   __restrict__ dstI,
                            const float* __restrict__ W1,
                            const __bf16* __restrict__ Wp,
                            float* __restrict__ acc,
                            int nTiles) {
  __shared__ float    W1s[64];
  __shared__ EdgeSmem sm[8];

  int tid = threadIdx.x;
  if (tid < 64) W1s[tid] = W1[tid];
  __syncthreads();   // single uniform barrier before any divergence

  int wave = tid >> 5;
  int lane = tid & 31;
  EdgeSmem* sw = &sm[wave];
  int gwave  = blockIdx.x * 8 + wave;
  int nWaves = gridDim.x * 8;
  int el    = lane & 15;      // edge-in-tile for phase1 / A-row
  int hi    = lane >> 4;      // lane half
  int ebase = hi * 8;         // D-layout row base -> edge offset
  const v16bf* Bp = (const v16bf*)Wp;

  for (int tile = gwave; tile < nTiles; tile += nWaves) {
    int eg = tile * 16 + el;
    int ec = (eg < N_EDGES) ? eg : (N_EDGES - 1);
    int s  = srcI[ec];

    // ---- phase 1: stage per-edge data into per-wave LDS ----
    if (!hi) {
      int d = dstI[ec];
      float px = pos[s*3+0], py = pos[s*3+1], pz = pos[s*3+2];
      float qx = pos[d*3+0], qy = pos[d*3+1], qz = pos[d*3+2];
      float vx = qx-px, vy = qy-py, vz = qz-pz;
      float l  = fmaxf(sqrtf(vx*vx + vy*vy + vz*vz), 1e-8f);
      float il = __builtin_amdgcn_rcpf(l);
      sw->len[el] = l;
      sw->dir[el][0] = vx*il; sw->dir[el][1] = vy*il; sw->dir[el][2] = vz*il;
      sw->dst[el] = d;
      #pragma unroll
      for (int u = 0; u < 8; u++) sw->xs[el][u] = x[s*32 + u];
    } else {
      #pragma unroll
      for (int u = 0; u < 8; u++) {
        sw->xv[el][u][0] = x[s*32 + 8 + u*3 + 0];
        sw->xv[el][u][1] = x[s*32 + 8 + u*3 + 1];
        sw->xv[el][u][2] = x[s*32 + 8 + u*3 + 2];
      }
    }
    __asm__ volatile("s_wait_dscnt 0\n\ts_wait_loadcnt 0" ::: "memory");

    // dot[e][u] = xv[e][u] . dir[e]   (each lane does 4 u's for its edge)
    {
      float dx = sw->dir[el][0], dy = sw->dir[el][1], dz = sw->dir[el][2];
      #pragma unroll
      for (int uu = 0; uu < 4; uu++) {
        int u = hi*4 + uu;
        sw->dot[el][u] = sw->xv[el][u][0]*dx + sw->xv[el][u][1]*dy + sw->xv[el][u][2]*dz;
      }
    }
    __asm__ volatile("s_wait_dscnt 0" ::: "memory");

    // ---- phase 2: build A (h = silu(len*W1)) in 16-bit A layout ----
    // lane half h: VGPR v<4 -> K=2v+j+8h ; v>=4 -> K=8+2v+j+8h ; M = lane%16
    float len_e = sw->len[el];
    v16bf A0, A1;
    #pragma unroll
    for (int v = 0; v < 8; v++) {
      #pragma unroll
      for (int j = 0; j < 2; j++) {
        int klo = (v < 4) ? (2*v + j + 8*hi) : (8 + 2*v + j + 8*hi);
        A0[2*v + j] = (__bf16)silu_f(len_e * W1s[klo]);
        A1[2*v + j] = (__bf16)silu_f(len_e * W1s[klo + 32]);
      }
    }

    v8f ms = {}, cC = {}, mv0 = {}, mv1 = {}, mv2 = {};

    // group A (tiles 0..7): pathA  -> m_s, coeff = xs[e][u]
    #pragma unroll
    for (int u = 0; u < 8; u++) {
      int t = u;
      v8f w = {};
      w = __builtin_amdgcn_wmma_f32_16x16x32_bf16(false, A0, false, Bp[(t*2+0)*32 + lane], (short)0, w, false, false);
      w = __builtin_amdgcn_wmma_f32_16x16x32_bf16(false, A1, false, Bp[(t*2+1)*32 + lane], (short)0, w, false, false);
      #pragma unroll
      for (int r = 0; r < 8; r++) ms[r] += sw->xs[ebase + r][u] * w[r];
    }
    // group B (tiles 8..15): pathB -> m_s, coeff = dot[e][u]
    #pragma unroll
    for (int u = 0; u < 8; u++) {
      int t = 8 + u;
      v8f w = {};
      w = __builtin_amdgcn_wmma_f32_16x16x32_bf16(false, A0, false, Bp[(t*2+0)*32 + lane], (short)0, w, false, false);
      w = __builtin_amdgcn_wmma_f32_16x16x32_bf16(false, A1, false, Bp[(t*2+1)*32 + lane], (short)0, w, false, false);
      #pragma unroll
      for (int r = 0; r < 8; r++) ms[r] += sw->dot[ebase + r][u] * w[r];
    }
    // group C (tiles 16..23): pathC coefficient -> cC (later * SQ3*dir)
    #pragma unroll
    for (int u = 0; u < 8; u++) {
      int t = 16 + u;
      v8f w = {};
      w = __builtin_amdgcn_wmma_f32_16x16x32_bf16(false, A0, false, Bp[(t*2+0)*32 + lane], (short)0, w, false, false);
      w = __builtin_amdgcn_wmma_f32_16x16x32_bf16(false, A1, false, Bp[(t*2+1)*32 + lane], (short)0, w, false, false);
      #pragma unroll
      for (int r = 0; r < 8; r++) cC[r] += sw->xs[ebase + r][u] * w[r];
    }
    // group D (tiles 24..31): pathD -> m_v, coeff = xv[e][u][i]
    #pragma unroll
    for (int u = 0; u < 8; u++) {
      int t = 24 + u;
      v8f w = {};
      w = __builtin_amdgcn_wmma_f32_16x16x32_bf16(false, A0, false, Bp[(t*2+0)*32 + lane], (short)0, w, false, false);
      w = __builtin_amdgcn_wmma_f32_16x16x32_bf16(false, A1, false, Bp[(t*2+1)*32 + lane], (short)0, w, false, false);
      #pragma unroll
      for (int r = 0; r < 8; r++) {
        mv0[r] += sw->xv[ebase + r][u][0] * w[r];
        mv1[r] += sw->xv[ebase + r][u][1] * w[r];
        mv2[r] += sw->xv[ebase + r][u][2] * w[r];
      }
    }

    // finish m_v: += cC * Y1  (Y1 = sqrt(3)*dir)
    const float SQ3 = 1.7320508075688772f;
    #pragma unroll
    for (int r = 0; r < 8; r++) {
      int et = ebase + r;
      float c = cC[r] * SQ3;
      mv0[r] += c * sw->dir[et][0];
      mv1[r] += c * sw->dir[et][1];
      mv2[r] += c * sw->dir[et][2];
    }

    // scatter-add into node accumulator (cols: [0..15]=m_s, [16..63]=m_v w*3+i)
    int n = el;  // D-layout col = lane%16
    #pragma unroll
    for (int r = 0; r < 8; r++) {
      int et = ebase + r;
      int eglob = tile * 16 + et;
      if (eglob < N_EDGES) {
        int d = sw->dst[et];
        float* o = acc + (long)d * 64;
        atomicAdd(o + n,               ms[r]);
        atomicAdd(o + 16 + n*3 + 0,    mv0[r]);
        atomicAdd(o + 16 + n*3 + 1,    mv1[r]);
        atomicAdd(o + 16 + n*3 + 2,    mv2[r]);
      }
    }
    __asm__ volatile("" ::: "memory");
  }
}

// ---------------------------------------------------------------------------
// Node kernel: one wave32 per 16-node tile, fp32 WMMA (16x16x4, K=16 -> 4 steps).
//   S  = os @ Ws, G = os @ Wg, NS_i = ov[:,:,i] @ Wns   (i = 0..2)
//   out = [ silu(S/4), sigmoid(G/4) * NS_i/4 ]
// 32-bit A 16x4 layout: lanes 0-15 M=lane, VGPR0=K0,VGPR1=K1; lanes 16-31 K=2,3.
// 32-bit B 4x16 layout: N=lane%16; lanes 0-15 K=0,1; lanes 16-31 K=2,3.
// ---------------------------------------------------------------------------
__launch_bounds__(256)
__global__ void node_kernel(const float* __restrict__ acc,
                            const float* __restrict__ Ws,
                            const float* __restrict__ Wns,
                            const float* __restrict__ Wg,
                            float* __restrict__ out, int nTiles) {
  int tid  = threadIdx.x;
  int wave = tid >> 5, lane = tid & 31;
  int tile = blockIdx.x * 8 + wave;
  if (tile >= nTiles) return;
  int n  = lane & 15;
  int hi = lane >> 4;
  int nodeA = tile * 16 + n;                       // A-matrix row for this lane
  int nodeC = (nodeA < N_NODES) ? nodeA : (N_NODES - 1);
  const float* arow = acc + (long)nodeC * 64;

  v8f accS = {}, accG = {}, accN0 = {}, accN1 = {}, accN2 = {};
  #pragma unroll
  for (int kk = 0; kk < 4; kk++) {
    int k0 = kk * 4 + 2 * hi;                      // first K this lane supplies
    v2f aS, a0, a1, a2, bS, bG, bN;
    aS.x = arow[k0];                aS.y = arow[k0 + 1];                 // os
    a0.x = arow[16 + k0*3 + 0];     a0.y = arow[16 + (k0+1)*3 + 0];     // ov_x
    a1.x = arow[16 + k0*3 + 1];     a1.y = arow[16 + (k0+1)*3 + 1];     // ov_y
    a2.x = arow[16 + k0*3 + 2];     a2.y = arow[16 + (k0+1)*3 + 2];     // ov_z
    bS.x = Ws [k0*16 + n];          bS.y = Ws [(k0+1)*16 + n];
    bG.x = Wg [k0*16 + n];          bG.y = Wg [(k0+1)*16 + n];
    bN.x = Wns[k0*16 + n];          bN.y = Wns[(k0+1)*16 + n];
    accS  = __builtin_amdgcn_wmma_f32_16x16x4_f32(false, aS, false, bS, (short)0, accS,  false, false);
    accG  = __builtin_amdgcn_wmma_f32_16x16x4_f32(false, aS, false, bG, (short)0, accG,  false, false);
    accN0 = __builtin_amdgcn_wmma_f32_16x16x4_f32(false, a0, false, bN, (short)0, accN0, false, false);
    accN1 = __builtin_amdgcn_wmma_f32_16x16x4_f32(false, a1, false, bN, (short)0, accN1, false, false);
    accN2 = __builtin_amdgcn_wmma_f32_16x16x4_f32(false, a2, false, bN, (short)0, accN2, false, false);
  }

  const float ln = 0.25f;  // 1/sqrt(MUL_OUT)
  #pragma unroll
  for (int r = 0; r < 8; r++) {
    int node = tile * 16 + hi * 8 + r;             // D-layout row -> node
    if (node < N_NODES) {
      float sv = silu_f(accS[r] * ln);
      float gv = sigm_f(accG[r] * ln);
      float* o = out + (long)node * 64;
      o[n]            = sv;
      o[16 + n*3 + 0] = gv * accN0[r] * ln;
      o[16 + n*3 + 1] = gv * accN1[r] * ln;
      o[16 + n*3 + 2] = gv * accN2[r] * ln;
    }
  }
}

// ---------------------------------------------------------------------------
extern "C" void kernel_launch(void* const* d_in, const int* in_sizes, int n_in,
                              void* d_out, int out_size, void* d_ws, size_t ws_size,
                              hipStream_t stream) {
  const float* x   = (const float*)d_in[0];
  const float* pos = (const float*)d_in[1];
  const int*   ei  = (const int*)  d_in[2];
  const float* W1  = (const float*)d_in[3];
  const float* W2  = (const float*)d_in[4];
  const float* Ws  = (const float*)d_in[5];
  const float* Wns = (const float*)d_in[6];
  const float* Wg  = (const float*)d_in[7];
  float* out = (float*)d_out;

  float*  acc = (float*)d_ws;                                       // N*64 f32
  __bf16* Wp  = (__bf16*)((char*)d_ws + (size_t)N_NODES*64*4);      // 64*512 bf16

  hipMemsetAsync(acc, 0, (size_t)N_NODES * 64 * sizeof(float), stream);
  prepack_w2<<<128, 256, 0, stream>>>(W2, Wp);

  int nTiles = (N_EDGES + 15) / 16;
  edge_kernel<<<640, 256, 0, stream>>>(x, pos, ei, ei + N_EDGES, W1, Wp, acc, nTiles);

  int nTilesN = (N_NODES + 15) / 16;
  node_kernel<<<(nTilesN + 7) / 8, 256, 0, stream>>>(acc, Ws, Wns, Wg, out, nTilesN);
}
